// Qwen2DecoderLayer_89550068121656
// MI455X (gfx1250) — matrix operand order repaired and checked by reading, compiled, not run on hardware
//
#include <hip/hip_runtime.h>
#include <hip/hip_bf16.h>

// ---------------- problem constants ----------------
constexpr int SEQ    = 2048;
constexpr int DMODEL = 2048;
constexpr int NH     = 16;
constexpr int NKV    = 4;
constexpr int HD     = 128;
constexpr int DFF    = 8192;

typedef __attribute__((ext_vector_type(16))) __bf16 v16bf;
typedef __attribute__((ext_vector_type(8)))  __bf16 v8bf;
typedef __attribute__((ext_vector_type(8)))  float  v8f;
typedef __attribute__((ext_vector_type(4)))  float  v4f;

typedef __attribute__((address_space(3))) unsigned char lds_byte;

// epilogue flags
#define FLAG_BF16OUT 1
#define FLAG_TRANS   2
#define FLAG_RES     4
#define FLAG_HEADS   8
#define FLAG_NT      16   // non-temporal fp32 C store (streams >> L2)

// CDNA5 async global->LDS copy (16B per lane), tracked with ASYNCcnt.
__device__ __forceinline__ void async_b128(unsigned ldsOff, const void* g)
{
    asm volatile("global_load_async_to_lds_b128 %0, %1, off"
                 :: "v"(ldsOff), "v"((unsigned long long)(uintptr_t)g)
                 : "memory");
}
__device__ __forceinline__ void wait_async_le2()
{
    asm volatile("s_wait_asynccnt 0x2" ::: "memory");
}
__device__ __forceinline__ void wait_async_0()
{
    asm volatile("s_wait_asynccnt 0x0" ::: "memory");
}

// fp32 weight row -> bf16 B fragment (32x16), non-temporal (single-use stream)
__device__ __forceinline__ v16bf load_bfrag_f32_nt(const float* pb)
{
    union { v16bf v; __bf16 e[16]; } ub;
    const v4f* q = reinterpret_cast<const v4f*>(pb);
    #pragma unroll
    for (int t = 0; t < 4; ++t) {
        v4f f = __builtin_nontemporal_load(q + t);
        ub.e[t * 4 + 0] = (__bf16)f.x;
        ub.e[t * 4 + 1] = (__bf16)f.y;
        ub.e[t * 4 + 2] = (__bf16)f.z;
        ub.e[t * 4 + 3] = (__bf16)f.w;
    }
    return ub.v;
}

// ---------------------------------------------------------------------------
// WMMA GEMM:  C[M,N] = scale * (A[M,K](bf16) @ B[N,K]^T) + bias[n]  (+res)
//   * block tile 128x128, 8 waves, each wave 32x64 (2x4 wmma tiles), K-step 32
//   * A tile staged in LDS via global_load_async_to_lds_b128, double-buffered
//   * B: fp32 weights (NT stream, cvt in-register) or bf16 activations
//   * blockIdx.z batches attention heads: A += z*aZ, B += (z/zdivB)*bZ, C += z*cZ
// ---------------------------------------------------------------------------
template <bool B_IS_F32>
__global__ __launch_bounds__(256) void gemm_wmma(
    const __bf16* __restrict__ A0, int lda, unsigned long long aZ,
    const void* __restrict__ Bp0, int ldb, unsigned long long bZ, int zdivB,
    const float* __restrict__ bias, float scale,
    void* __restrict__ Cp0, int ldc, unsigned long long cZ,
    const float* __restrict__ res, int ldr,
    int M, int N, int K, int flags)
{
    constexpr int APITCH = 80;            // 64B of K data + pad (16B-aligned, bank-spread)
    __shared__ __align__(16) unsigned char smA[2][128 * APITCH];

    const int z = blockIdx.z;
    const __bf16* A = A0 + (size_t)z * aZ;
    const char*   B = (const char*)Bp0 + (size_t)(z / zdivB) * bZ * (B_IS_F32 ? 4 : 2);
    char*        Cp = (char*)Cp0 + (size_t)z * cZ * ((flags & FLAG_BF16OUT) ? 2 : 4);

    const int bm = blockIdx.y * 128;
    const int bn = blockIdx.x * 128;

    const int lane   = threadIdx.x & 31;
    const int wave   = threadIdx.x >> 5;
    const int half   = lane >> 4;          // 0: lanes 0-15, 1: lanes 16-31
    const int laneLo = lane & 15;
    const int wmLoc  = (wave >> 1) * 32;   // wave row within block tile
    const int wn     = bn + (wave & 1) * 64;

    const unsigned ldsA = (unsigned)(uintptr_t)(lds_byte*)&smA[0][0];

    // stage one 128x32 bf16 A tile into LDS buffer `buf` (2 async b128 per thread)
    const int cpRow  = threadIdx.x >> 1;   // 0..127
    const int cpHalf = threadIdx.x & 1;    // 0..1 -> 16-element (32B) span
    auto stageA = [&](int buf, int k0) {
        const __bf16* g = A + (size_t)(bm + cpRow) * lda + (k0 + cpHalf * 16);
        const unsigned l = ldsA + (unsigned)(buf * (128 * APITCH) + cpRow * APITCH + cpHalf * 32);
        async_b128(l, g);
        async_b128(l + 16, g + 8);
    };

    v8f c[2][4] = {};

    const int nk = K >> 5;
    stageA(0, 0);

    for (int i = 0; i < nk; ++i) {
        if (i + 1 < nk) {
            stageA((i + 1) & 1, (i + 1) * 32);
            wait_async_le2();              // in-order returns => tile i complete
        } else {
            wait_async_0();
        }
        __syncthreads();

        const int k0 = i * 32;
        const unsigned char* bufA = &smA[i & 1][0];

        // A fragments (16x32) from LDS: elems 0..15 = 16 contiguous K from half*16
        v16bf a[2];
        #pragma unroll
        for (int ii = 0; ii < 2; ++ii) {
            const unsigned char* pa = bufA + (wmLoc + ii * 16 + laneLo) * APITCH + half * 32;
            union { v16bf v; v8bf h[2]; } ua;
            ua.h[0] = *reinterpret_cast<const v8bf*>(pa);
            ua.h[1] = *reinterpret_cast<const v8bf*>(pa + 16);
            a[ii] = ua.v;
        }

        // B fragments (32x16): lane holds column, 16 contiguous K from half*16
        v16bf b[4];
        #pragma unroll
        for (int j = 0; j < 4; ++j) {
            const int row = wn + j * 16 + laneLo;
            if constexpr (B_IS_F32) {
                b[j] = load_bfrag_f32_nt((const float*)B + (size_t)row * ldb + (k0 + half * 16));
            } else {
                const __bf16* pb = (const __bf16*)B + (size_t)row * ldb + (k0 + half * 16);
                union { v16bf v; v8bf h[2]; } ub;
                ub.h[0] = *reinterpret_cast<const v8bf*>(pb);
                ub.h[1] = *reinterpret_cast<const v8bf*>(pb + 8);
                b[j] = ub.v;
            }
        }

        #pragma unroll
        for (int ii = 0; ii < 2; ++ii)
            #pragma unroll
            for (int j = 0; j < 4; ++j)
                c[ii][j] = __builtin_amdgcn_wmma_f32_16x16x32_bf16(
                    false, a[ii], false, b[j], (short)0, c[ii][j], false, false);

        __syncthreads();                   // protect LDS buf before next overwrite
    }

    // Epilogue. C/D layout: lanes 0-15 -> M=r, lanes 16-31 -> M=8+r; N=lane&15
    const int mOff = half * 8;
    const int wm = bm + wmLoc;
    #pragma unroll
    for (int ii = 0; ii < 2; ++ii) {
        #pragma unroll
        for (int j = 0; j < 4; ++j) {
            const int n = wn + j * 16 + laneLo;
            #pragma unroll
            for (int r = 0; r < 8; ++r) {
                const int m = wm + ii * 16 + mOff + r;
                float v = c[ii][j][r] * scale;
                if (bias) v += bias[n];
                if (flags & FLAG_RES) v += res[(size_t)m * ldr + n];
                if (flags & FLAG_BF16OUT) {
                    __bf16* C = (__bf16*)Cp;
                    if (flags & FLAG_TRANS)
                        C[(size_t)n * ldc + m] = (__bf16)v;                         // vT: [N,M]
                    else if (flags & FLAG_HEADS)
                        C[((size_t)(n >> 7) * M + m) * HD + (n & (HD - 1))] = (__bf16)v; // [h,S,hd]
                    else
                        C[(size_t)m * ldc + n] = (__bf16)v;
                } else {
                    float* C = (float*)Cp;
                    if (flags & FLAG_NT)
                        __builtin_nontemporal_store(v, &C[(size_t)m * ldc + n]);
                    else
                        C[(size_t)m * ldc + n] = v;
                }
            }
        }
    }
}

// ---------------------------------------------------------------------------
// Fused MLP gate/up:  T = gelu_erf(A @ Wg^T) * (A @ Wu^T), bf16 out.
// Same tiling/staging as gemm_wmma; two accumulator sets, 16 wmma per K-step.
// ---------------------------------------------------------------------------
__global__ __launch_bounds__(256) void mlp_gateup_wmma(
    const __bf16* __restrict__ A, const float* __restrict__ Wg,
    const float* __restrict__ Wu, __bf16* __restrict__ T)
{
    constexpr int APITCH = 80;
    __shared__ __align__(16) unsigned char smA[2][128 * APITCH];

    const int bm = blockIdx.y * 128;
    const int bn = blockIdx.x * 128;

    const int lane   = threadIdx.x & 31;
    const int wave   = threadIdx.x >> 5;
    const int half   = lane >> 4;
    const int laneLo = lane & 15;
    const int wmLoc  = (wave >> 1) * 32;
    const int wn     = bn + (wave & 1) * 64;

    const unsigned ldsA = (unsigned)(uintptr_t)(lds_byte*)&smA[0][0];
    const int cpRow  = threadIdx.x >> 1;
    const int cpHalf = threadIdx.x & 1;
    auto stageA = [&](int buf, int k0) {
        const __bf16* g = A + (size_t)(bm + cpRow) * DMODEL + (k0 + cpHalf * 16);
        const unsigned l = ldsA + (unsigned)(buf * (128 * APITCH) + cpRow * APITCH + cpHalf * 32);
        async_b128(l, g);
        async_b128(l + 16, g + 8);
    };

    v8f cg[2][4] = {};
    v8f cu[2][4] = {};

    constexpr int nk = DMODEL >> 5;
    stageA(0, 0);

    for (int i = 0; i < nk; ++i) {
        if (i + 1 < nk) {
            stageA((i + 1) & 1, (i + 1) * 32);
            wait_async_le2();
        } else {
            wait_async_0();
        }
        __syncthreads();

        const int k0 = i * 32;
        const unsigned char* bufA = &smA[i & 1][0];

        v16bf a[2];
        #pragma unroll
        for (int ii = 0; ii < 2; ++ii) {
            const unsigned char* pa = bufA + (wmLoc + ii * 16 + laneLo) * APITCH + half * 32;
            union { v16bf v; v8bf h[2]; } ua;
            ua.h[0] = *reinterpret_cast<const v8bf*>(pa);
            ua.h[1] = *reinterpret_cast<const v8bf*>(pa + 16);
            a[ii] = ua.v;
        }

        #pragma unroll
        for (int j = 0; j < 4; ++j) {
            const size_t boff = (size_t)(wn + j * 16 + laneLo) * DMODEL + (k0 + half * 16);
            v16bf bg = load_bfrag_f32_nt(Wg + boff);
            #pragma unroll
            for (int ii = 0; ii < 2; ++ii)
                cg[ii][j] = __builtin_amdgcn_wmma_f32_16x16x32_bf16(
                    false, a[ii], false, bg, (short)0, cg[ii][j], false, false);
            v16bf bu = load_bfrag_f32_nt(Wu + boff);
            #pragma unroll
            for (int ii = 0; ii < 2; ++ii)
                cu[ii][j] = __builtin_amdgcn_wmma_f32_16x16x32_bf16(
                    false, a[ii], false, bu, (short)0, cu[ii][j], false, false);
        }

        __syncthreads();
    }

    const int mOff = half * 8;
    const int wm = bm + wmLoc;
    #pragma unroll
    for (int ii = 0; ii < 2; ++ii) {
        #pragma unroll
        for (int j = 0; j < 4; ++j) {
            const int n = wn + j * 16 + laneLo;
            #pragma unroll
            for (int r = 0; r < 8; ++r) {
                const int m = wm + ii * 16 + mOff + r;
                const float g = cg[ii][j][r];
                const float u = cu[ii][j][r];
                const float ge = 0.5f * g * (1.f + erff(g * 0.70710678118654752f));
                T[(size_t)m * DFF + n] = (__bf16)(ge * u);
            }
        }
    }
}

// ---------------------------------------------------------------------------
__global__ __launch_bounds__(256) void rmsnorm_kernel(
    const float* __restrict__ x, const float* __restrict__ w,
    __bf16* __restrict__ out)
{
    __shared__ float red[256];
    const int row = blockIdx.x, tid = threadIdx.x;
    const float* xr = x + (size_t)row * DMODEL;
    float ss = 0.f;
    for (int i = tid; i < DMODEL; i += 256) { float v = xr[i]; ss += v * v; }
    red[tid] = ss; __syncthreads();
    for (int s = 128; s > 0; s >>= 1) {
        if (tid < s) red[tid] += red[tid + s];
        __syncthreads();
    }
    const float inv = rsqrtf(red[0] / (float)DMODEL + 1e-6f);
    __bf16* orow = out + (size_t)row * DMODEL;
    for (int i = tid; i < DMODEL; i += 256) orow[i] = (__bf16)(w[i] * xr[i] * inv);
}

// one block per (head,row); row length SEQ = 2048 = 8 x 256.
// Scores stream (268 MB) >> L2: read with NT loads.
__global__ __launch_bounds__(256) void softmax_kernel(
    const float* __restrict__ scores, __bf16* __restrict__ probs)
{
    __shared__ float red[256];
    const size_t row = blockIdx.x;
    const float* sr = scores + row * SEQ;
    __bf16* pr = probs + row * SEQ;
    const int tid = threadIdx.x;
    float v[8];
    float mx = -3.402823466e38f;
    #pragma unroll
    for (int i = 0; i < 8; ++i) {
        v[i] = __builtin_nontemporal_load(sr + tid + i * 256);
        mx = fmaxf(mx, v[i]);
    }
    red[tid] = mx; __syncthreads();
    for (int s = 128; s > 0; s >>= 1) {
        if (tid < s) red[tid] = fmaxf(red[tid], red[tid + s]);
        __syncthreads();
    }
    mx = red[0]; __syncthreads();
    float sum = 0.f;
    #pragma unroll
    for (int i = 0; i < 8; ++i) { v[i] = __expf(v[i] - mx); sum += v[i]; }
    red[tid] = sum; __syncthreads();
    for (int s = 128; s > 0; s >>= 1) {
        if (tid < s) red[tid] += red[tid + s];
        __syncthreads();
    }
    const float inv = 1.f / red[0];
    #pragma unroll
    for (int i = 0; i < 8; ++i) pr[tid + i * 256] = (__bf16)(v[i] * inv);
}

// ---------------------------------------------------------------------------
extern "C" void kernel_launch(void* const* d_in, const int* in_sizes, int n_in,
                              void* d_out, int out_size, void* d_ws, size_t ws_size,
                              hipStream_t stream)
{
    (void)in_sizes; (void)n_in; (void)out_size; (void)ws_size;

    const float* hidden = (const float*)d_in[0];
    const float* q_w    = (const float*)d_in[1];
    const float* q_b    = (const float*)d_in[2];
    const float* k_w    = (const float*)d_in[3];
    const float* k_b    = (const float*)d_in[4];
    const float* v_w    = (const float*)d_in[5];
    const float* v_b    = (const float*)d_in[6];
    const float* o_w    = (const float*)d_in[7];
    const float* gate_w = (const float*)d_in[8];
    const float* up_w   = (const float*)d_in[9];
    const float* down_w = (const float*)d_in[10];
    const float* ln1_w  = (const float*)d_in[11];
    const float* ln2_w  = (const float*)d_in[12];

    float* out    = (float*)d_out;                       // [SEQ, DMODEL]
    float* scores = out + (size_t)SEQ * DMODEL;          // [NH, SEQ, SEQ] pre-softmax

    char* p = (char*)d_ws;
    auto alloc = [&](size_t bytes) -> char* {
        char* r = p;
        p += (bytes + 255) & ~(size_t)255;
        return r;
    };
    __bf16* x_bf  = (__bf16*)alloc((size_t)SEQ * DMODEL * 2);   // rmsnorm1(x)
    __bf16* q_bf  = (__bf16*)alloc((size_t)NH  * SEQ * HD * 2); // [h, S, hd]
    __bf16* k_bf  = (__bf16*)alloc((size_t)NKV * SEQ * HD * 2); // [kv, S, hd]
    __bf16* vT    = (__bf16*)alloc((size_t)NKV * HD * SEQ * 2); // [kv*hd, S]
    __bf16* probs = (__bf16*)alloc((size_t)NH  * SEQ * SEQ * 2);
    __bf16* ctx   = (__bf16*)alloc((size_t)SEQ * NH * HD * 2);  // [S, 2048]
    float*  hbuf  = (float*) alloc((size_t)SEQ * DMODEL * 4);   // residual + attn
    __bf16* y_bf  = (__bf16*)alloc((size_t)SEQ * DMODEL * 2);   // rmsnorm2
    __bf16* t_bf  = (__bf16*)alloc((size_t)SEQ * DFF * 2);      // gelu(gate)*up

    const dim3 blk(256);
    const float scale = 0.08838834764831845f;  // HEAD_DIM^-0.5

    // 1. x = rmsnorm(hidden, ln1)
    rmsnorm_kernel<<<SEQ, 256, 0, stream>>>(hidden, ln1_w, x_bf);

    // 2. Q/K/V projections (weights streamed fp32 NT -> bf16 in-register)
    gemm_wmma<true><<<dim3(DMODEL / 128, SEQ / 128, 1), blk, 0, stream>>>(
        x_bf, DMODEL, 0, q_w, DMODEL, 0, 1, q_b, 1.f,
        q_bf, 0, 0, nullptr, 0, SEQ, NH * HD, DMODEL, FLAG_BF16OUT | FLAG_HEADS);
    gemm_wmma<true><<<dim3((NKV * HD) / 128, SEQ / 128, 1), blk, 0, stream>>>(
        x_bf, DMODEL, 0, k_w, DMODEL, 0, 1, k_b, 1.f,
        k_bf, 0, 0, nullptr, 0, SEQ, NKV * HD, DMODEL, FLAG_BF16OUT | FLAG_HEADS);
    gemm_wmma<true><<<dim3((NKV * HD) / 128, SEQ / 128, 1), blk, 0, stream>>>(
        x_bf, DMODEL, 0, v_w, DMODEL, 0, 1, v_b, 1.f,
        vT, SEQ, 0, nullptr, 0, SEQ, NKV * HD, DMODEL, FLAG_BF16OUT | FLAG_TRANS);

    // 3. scores = scale * Q @ K^T -> fp32 NT stores into d_out (2nd output); z = head
    gemm_wmma<false><<<dim3(SEQ / 128, SEQ / 128, NH), blk, 0, stream>>>(
        q_bf, HD, (size_t)SEQ * HD,
        k_bf, HD, (size_t)SEQ * HD, 4,
        nullptr, scale,
        scores, SEQ, (size_t)SEQ * SEQ, nullptr, 0,
        SEQ, SEQ, HD, FLAG_NT);

    // 4. probs = softmax(scores)
    softmax_kernel<<<NH * SEQ, 256, 0, stream>>>(scores, probs);

    // 5. ctx[:, h*128:(h+1)*128] = probs_h @ V_h ; z = head
    gemm_wmma<false><<<dim3(HD / 128, SEQ / 128, NH), blk, 0, stream>>>(
        probs, SEQ, (size_t)SEQ * SEQ,
        vT, SEQ, (size_t)HD * SEQ, 4,
        nullptr, 1.f,
        ctx, NH * HD, (size_t)HD, nullptr, 0,
        SEQ, HD, SEQ, FLAG_BF16OUT);

    // 6. hbuf = hidden + ctx @ o_w^T   (residual fused in epilogue)
    gemm_wmma<true><<<dim3(DMODEL / 128, SEQ / 128, 1), blk, 0, stream>>>(
        ctx, NH * HD, 0, o_w, NH * HD, 0, 1, nullptr, 1.f,
        hbuf, DMODEL, 0, hidden, DMODEL, SEQ, DMODEL, NH * HD, FLAG_RES);

    // 7. y = rmsnorm(hbuf, ln2)
    rmsnorm_kernel<<<SEQ, 256, 0, stream>>>(hbuf, ln2_w, y_bf);

    // 8. MLP: fused gate/up GEMM pair + exact-erf GELU*up epilogue
    mlp_gateup_wmma<<<dim3(DFF / 128, SEQ / 128, 1), blk, 0, stream>>>(
        y_bf, gate_w, up_w, t_bf);

    // 9. out = hbuf + t @ down_w^T   (residual fused)
    gemm_wmma<true><<<dim3(DMODEL / 128, SEQ / 128, 1), blk, 0, stream>>>(
        t_bf, DFF, 0, down_w, DFF, 0, 1, nullptr, 1.f,
        out, DMODEL, 0, hbuf, DMODEL, SEQ, DMODEL, DFF, FLAG_RES);
}
